// MultiHeadSelfAttention_61607010894422
// MI455X (gfx1250) — matrix operand order repaired and checked by reading, not compiled, not run on hardware
//
#include <hip/hip_runtime.h>
#include <math.h>

// ---------------------------------------------------------------------------
// MI455X (gfx1250) multi-head self-attention forward.
// Roofline: ~60 GFLOP vs ~50MB HBM traffic -> compute bound. Use f16 WMMA
// (v_wmma_f32_16x16x32_f16) with f32 accumulation. All intermediates fit in
// the 192MB L2, so GEMMs stream fragments straight from L2 (no LDS staging);
// attention uses per-wave LDS only to reshape P from C-layout to A-layout.
// ---------------------------------------------------------------------------

typedef __attribute__((ext_vector_type(16))) _Float16 v16h;
typedef __attribute__((ext_vector_type(8)))  _Float16 v8h;
typedef __attribute__((ext_vector_type(8)))  float    v8f;

constexpr int Bn  = 2;
constexpr int Sn  = 2048;
constexpr int Dn  = 1024;   // D_IN
constexpr int Hn  = 16;
constexpr int DHn = 64;
constexpr int DM  = 1024;   // D_MODEL
constexpr float NEGBIG = 1.0e10f;

// A-fragment (16x32 f16): lane L holds row M=L%16.
//   lanes 0-15 : halves 0..15 = K k0+0..15
//   lanes16-31 : halves 0..15 = K k0+16..31
// -> two 16B loads at p and p+8, where p already includes +16 for upper half.
__device__ inline v16h load_a_frag(const _Float16* p) {
  union { v16h v; v8h h[2]; } u;
  __builtin_memcpy(&u.h[0], p,      16);
  __builtin_memcpy(&u.h[1], p + 8, 16);
  return u.v;
}

// B-fragment (32x16 f16): lane L holds column N=L%16.
//   lanes 0-15 : halves = K 0..15 ; lanes 16-31 : halves = K 16..31
// -> one contiguous 32B load (caller adds +16 halves for upper half-wave).
__device__ inline v16h load_b_frag(const _Float16* p) {
  v16h v;
  __builtin_memcpy(&v, p, 32);
  return v;
}

__device__ inline v8f wmma_f16(v16h a, v16h b, v8f c) {
  return __builtin_amdgcn_wmma_f32_16x16x32_f16(false, a, false, b,
                                                (short)0, c, false, false);
}

// ---------------------------------------------------------------------------
// Kernel 1: LayerNorm, f32 in -> f16 out. One block per row (1024 elems).
// ---------------------------------------------------------------------------
__global__ __launch_bounds__(256)
void ln_kernel(const float* __restrict__ x, const float* __restrict__ lnw,
               const float* __restrict__ lnb, _Float16* __restrict__ xn) {
  __shared__ float red[16];
  const int row  = blockIdx.x;
  const int lane = threadIdx.x & 31;
  const int wv   = threadIdx.x >> 5;

  float4 v;
  __builtin_memcpy(&v, x + row * Dn + threadIdx.x * 4, 16);

  float s = v.x + v.y + v.z + v.w;
#pragma unroll
  for (int m = 16; m >= 1; m >>= 1) s += __shfl_xor(s, m, 32);
  if (lane == 0) red[wv] = s;
  __syncthreads();
  float mu = 0.f;
#pragma unroll
  for (int i = 0; i < 8; ++i) mu += red[i];
  mu *= (1.0f / Dn);

  float dx = v.x - mu, dy = v.y - mu, dz = v.z - mu, dw = v.w - mu;
  float q = dx * dx + dy * dy + dz * dz + dw * dw;
#pragma unroll
  for (int m = 16; m >= 1; m >>= 1) q += __shfl_xor(q, m, 32);
  if (lane == 0) red[8 + wv] = q;
  __syncthreads();
  float var = 0.f;
#pragma unroll
  for (int i = 0; i < 8; ++i) var += red[8 + i];
  var *= (1.0f / Dn);
  const float inv = rsqrtf(var + 1e-5f);

  float4 w4, b4;
  __builtin_memcpy(&w4, lnw + threadIdx.x * 4, 16);
  __builtin_memcpy(&b4, lnb + threadIdx.x * 4, 16);
  v8h out;  // only 4 used; pack 4 halves
  _Float16 o0 = (_Float16)(dx * inv * w4.x + b4.x);
  _Float16 o1 = (_Float16)(dy * inv * w4.y + b4.y);
  _Float16 o2 = (_Float16)(dz * inv * w4.z + b4.z);
  _Float16 o3 = (_Float16)(dw * inv * w4.w + b4.w);
  _Float16 tmp[4] = {o0, o1, o2, o3};
  __builtin_memcpy(xn + row * Dn + threadIdx.x * 4, tmp, 8);
  (void)out;
}

// ---------------------------------------------------------------------------
// Kernel 2: f32 -> f16 conversion (weights).
// ---------------------------------------------------------------------------
__global__ __launch_bounds__(256)
void cvt_kernel(const float* __restrict__ in, _Float16* __restrict__ out, int n) {
  int i = blockIdx.x * blockDim.x + threadIdx.x;
  if (i < n) out[i] = (_Float16)in[i];
}

// ---------------------------------------------------------------------------
// Kernel 3: projection GEMM  y[n,m] = sum_k Xn[n,k] * W[m,k] + bias[m]
// Block = 128 threads = 4 waves (2x2), each wave owns a 64x64 tile.
// transposeV==0: write (B,H,S,DH) f16 (for Q,K). ==1: write (B,H,DH,S) (V^T).
// ---------------------------------------------------------------------------
__global__ __launch_bounds__(128)
void proj_gemm(const _Float16* __restrict__ Xn,   // [B*S][Dn]
               const _Float16* __restrict__ W16,  // [DM][Dn]
               const float* __restrict__ bias,    // [DM]
               _Float16* __restrict__ out,
               int transposeV) {
  const int lane = threadIdx.x & 31;
  const int wv   = threadIdx.x >> 5;
  const int hl   = lane >> 4;      // half-wave index
  const int l16  = lane & 15;
  const int c0a  = hl * 16;        // A-fragment K shift
  const int c0b  = hl * 16;        // B-fragment K shift

  const int rowbase = blockIdx.x * 128 + (wv >> 1) * 64;
  const int colbase = blockIdx.y * 128 + (wv & 1) * 64;

  v8f acc[4][4] = {};
  for (int k0 = 0; k0 < Dn; k0 += 32) {
    v16h a[4], bf[4];
#pragma unroll
    for (int i = 0; i < 4; ++i)
      a[i] = load_a_frag(Xn + (rowbase + i * 16 + l16) * Dn + k0 + c0a);
#pragma unroll
    for (int j = 0; j < 4; ++j)
      bf[j] = load_b_frag(W16 + (colbase + j * 16 + l16) * Dn + k0 + c0b);
#pragma unroll
    for (int i = 0; i < 4; ++i)
#pragma unroll
      for (int j = 0; j < 4; ++j)
        acc[i][j] = wmma_f16(a[i], bf[j], acc[i][j]);
  }

#pragma unroll
  for (int i = 0; i < 4; ++i) {
#pragma unroll
    for (int j = 0; j < 4; ++j) {
      const int m  = colbase + j * 16 + l16;     // feature column
      const int h  = m >> 6;                     // head
      const int d  = m & 63;                     // dim in head
      const float bvv = bias[m];
      const int n0 = rowbase + i * 16 + hl * 8;  // first row this lane owns
      const int bb = n0 / Sn;
      const int s0 = n0 % Sn;                    // 16-row tiles never cross batch
      if (!transposeV) {
#pragma unroll
        for (int r = 0; r < 8; ++r) {
          _Float16 hv = (_Float16)(acc[i][j][r] + bvv);
          out[(((bb * Hn + h) * Sn) + (s0 + r)) * DHn + d] = hv;
        }
      } else {
        _Float16 pk[8];
#pragma unroll
        for (int r = 0; r < 8; ++r) pk[r] = (_Float16)(acc[i][j][r] + bvv);
        __builtin_memcpy(out + ((bb * Hn + h) * DHn + d) * Sn + s0, pk, 16);
      }
    }
  }
}

// ---------------------------------------------------------------------------
// Kernel 4: flash attention. Block = 4 waves; wave owns 16 q-rows; kv step 32.
// ---------------------------------------------------------------------------
__global__ __launch_bounds__(128)
void attn_kernel(const _Float16* __restrict__ Q,   // [B*H][S][DH]
                 const _Float16* __restrict__ K,   // [B*H][S][DH]
                 const _Float16* __restrict__ Vt,  // [B*H][DH][S]
                 const float* __restrict__ mask,   // [B][S]
                 float* __restrict__ out) {        // [B][S][DM]
  __shared__ _Float16 plds[4][16 * 32];            // per-wave P tile

  const int lane = threadIdx.x & 31;
  const int wv   = threadIdx.x >> 5;
  const int hl   = lane >> 4;
  const int l16  = lane & 15;
  const int c0a  = hl * 16;
  const int c0b  = hl * 16;

  const int bh = blockIdx.y;            // b*H + h
  const int bb = bh / Hn;
  const int hh = bh % Hn;
  const int q0 = blockIdx.x * 64 + wv * 16;

  // Q A-fragments (d = 0..31 and 32..63)
  const _Float16* qp = Q + ((long)bh * Sn + q0 + l16) * DHn;
  const v16h qa0 = load_a_frag(qp + c0a);
  const v16h qa1 = load_a_frag(qp + 32 + c0a);

  v8f o[4] = {};
  float mrun[8], lrun[8];
#pragma unroll
  for (int r = 0; r < 8; ++r) { mrun[r] = -__builtin_inff(); lrun[r] = 0.f; }

  const float scale = 0.125f;           // 1/sqrt(DH)
  const float l2e   = 1.44269504088896f;
  _Float16* pt = plds[wv];

  for (int kv0 = 0; kv0 < Sn; kv0 += 32) {
    // ---- scores: two 16x16 tiles over kv columns [0,16) and [16,32)
    v8f sc[2];
#pragma unroll
    for (int t = 0; t < 2; ++t) {
      const _Float16* kp = K + ((long)bh * Sn + kv0 + t * 16 + l16) * DHn;
      v16h kb0 = load_b_frag(kp + c0b);
      v16h kb1 = load_b_frag(kp + 32 + c0b);
      v8f a = {};
      a = wmma_f16(qa0, kb0, a);
      a = wmma_f16(qa1, kb1, a);
      sc[t] = a;
    }
    const float mv0 = mask[bb * Sn + kv0 + l16];
    const float mv1 = mask[bb * Sn + kv0 + 16 + l16];
    const float pen0 = (1.f - mv0) * NEGBIG;
    const float pen1 = (1.f - mv1) * NEGBIG;
#pragma unroll
    for (int r = 0; r < 8; ++r) {
      sc[0][r] = sc[0][r] * scale - pen0;
      sc[1][r] = sc[1][r] * scale - pen1;
    }

    // ---- online softmax: rows live across 16 lanes of each half-wave
    float mnew[8], sf[8];
#pragma unroll
    for (int r = 0; r < 8; ++r) {
      float t = fmaxf(sc[0][r], sc[1][r]);
#pragma unroll
      for (int m = 8; m >= 1; m >>= 1) t = fmaxf(t, __shfl_xor(t, m, 32));
      mnew[r] = fmaxf(mrun[r], t);
      sf[r]   = exp2f((mrun[r] - mnew[r]) * l2e);
    }
#pragma unroll
    for (int r = 0; r < 8; ++r) {
      const float p0 = exp2f((sc[0][r] - mnew[r]) * l2e);
      const float p1 = exp2f((sc[1][r] - mnew[r]) * l2e);
      float rs = p0 + p1;                 // denominator uses pre-mask exp
#pragma unroll
      for (int m = 8; m >= 1; m >>= 1) rs += __shfl_xor(rs, m, 32);
      lrun[r] = lrun[r] * sf[r] + rs;
      mrun[r] = mnew[r];
      const int row = r + hl * 8;         // C-layout row for this lane/slot
      _Float16 h0 = (_Float16)(p0 * mv0); // numerator gets attn *= mask
      _Float16 h1 = (_Float16)(p1 * mv1);
      __builtin_memcpy(pt + row * 32 + l16,      &h0, 2);
      __builtin_memcpy(pt + row * 32 + 16 + l16, &h1, 2);
    }
    // rescale O accumulators
#pragma unroll
    for (int nd = 0; nd < 4; ++nd)
#pragma unroll
      for (int r = 0; r < 8; ++r) o[nd][r] *= sf[r];

    // ---- P (A-layout from LDS) x V^T-fragments; DS ops are wave-ordered
    const v16h pa = load_a_frag(pt + l16 * 32 + c0a);
#pragma unroll
    for (int nd = 0; nd < 4; ++nd) {
      const _Float16* vp = Vt + ((long)bh * DHn + nd * 16 + l16) * Sn + kv0 + c0b;
      v16h vb = load_b_frag(vp);
      o[nd] = wmma_f16(pa, vb, o[nd]);
    }
  }

  // ---- normalize and write f32 output (B, S, D_MODEL)
#pragma unroll
  for (int r = 0; r < 8; ++r) {
    const float inv = lrun[r] > 0.f ? 1.0f / lrun[r] : 0.f;
    const int s = q0 + hl * 8 + r;
    float* op = out + ((long)bb * Sn + s) * DM + hh * DHn;
#pragma unroll
    for (int nd = 0; nd < 4; ++nd) op[nd * 16 + l16] = o[nd][r] * inv;
  }
}

// ---------------------------------------------------------------------------
// Host launcher
// ---------------------------------------------------------------------------
extern "C" void kernel_launch(void* const* d_in, const int* in_sizes, int n_in,
                              void* d_out, int out_size, void* d_ws, size_t ws_size,
                              hipStream_t stream) {
  const float* x    = (const float*)d_in[0];
  const float* mask = (const float*)d_in[1];
  const float* Wq   = (const float*)d_in[2];
  const float* bq   = (const float*)d_in[3];
  const float* Wk   = (const float*)d_in[4];
  const float* bk   = (const float*)d_in[5];
  const float* Wv   = (const float*)d_in[6];
  const float* bv   = (const float*)d_in[7];
  const float* lnw  = (const float*)d_in[8];
  const float* lnb  = (const float*)d_in[9];
  float* out = (float*)d_out;

  char* ws = (char*)d_ws;
  _Float16* Xn  = (_Float16*)(ws);                        //  8 MB: [4096][1024]
  _Float16* W16 = (_Float16*)(ws + (8u  << 20));          //  6 MB: 3x[1024][1024]
  _Float16* Qb  = (_Float16*)(ws + (14u << 20));          //  8 MB: [B*H][S][DH]
  _Float16* Kb  = (_Float16*)(ws + (22u << 20));          //  8 MB
  _Float16* Vt  = (_Float16*)(ws + (30u << 20));          //  8 MB: [B*H][DH][S]
  _Float16* Wq16 = W16;
  _Float16* Wk16 = W16 + DM * Dn;
  _Float16* Wv16 = W16 + 2 * DM * Dn;

  // 1) LayerNorm -> f16
  ln_kernel<<<dim3(Bn * Sn), 256, 0, stream>>>(x, lnw, lnb, Xn);

  // 2) weight conversion f32 -> f16
  const int nw = DM * Dn;
  cvt_kernel<<<dim3(nw / 256), 256, 0, stream>>>(Wq, Wq16, nw);
  cvt_kernel<<<dim3(nw / 256), 256, 0, stream>>>(Wk, Wk16, nw);
  cvt_kernel<<<dim3(nw / 256), 256, 0, stream>>>(Wv, Wv16, nw);

  // 3) Q/K/V projection GEMMs (V written transposed for the P·V fragments)
  dim3 ggrid((Bn * Sn) / 128, DM / 128);
  proj_gemm<<<ggrid, 128, 0, stream>>>(Xn, Wq16, bq, Qb, 0);
  proj_gemm<<<ggrid, 128, 0, stream>>>(Xn, Wk16, bk, Kb, 0);
  proj_gemm<<<ggrid, 128, 0, stream>>>(Xn, Wv16, bv, Vt, 1);

  // 4) flash attention
  attn_kernel<<<dim3(Sn / 64, Bn * Hn), 128, 0, stream>>>(Qb, Kb, Vt, mask, out);
}
